// LSTMGenerator_46059229283123
// MI455X (gfx1250) — matrix operand order repaired and checked by reading, not compiled, not run on hardware
//
#include <hip/hip_runtime.h>
#include <hip/hip_bf16.h>
#include <stdint.h>

// ---------------- problem constants ----------------
constexpr int kB   = 32;      // batch
constexpr int kH   = 1024;    // hidden
constexpr int kV   = 32000;   // vocab
constexpr int kL   = 128;     // sequence length
constexpr int kZD  = 128;     // latent dim
constexpr int kG   = 4096;    // 4*H (gates)
constexpr int kKC  = 32;      // kH / 32 K-chunks per GEMM
constexpr int kNB  = 500;     // kV / 64 logits blocks
constexpr int kBOS = 2;
constexpr int kAP  = 2 * kKC * 32 * 16;   // packed A-fragment elems (2 mtiles) = 32768

typedef __bf16 bf16;
typedef __bf16 v16bf __attribute__((ext_vector_type(16)));
typedef float  v8f   __attribute__((ext_vector_type(8)));

__device__ __forceinline__ float sigf(float x){ return 1.f / (1.f + __expf(-x)); }

__device__ __forceinline__ float gumbelf(unsigned x){
  x ^= x >> 17; x *= 0xed5ad4bbu; x ^= x >> 11; x *= 0xac4c1b51u;
  x ^= x >> 15; x *= 0x31848babu; x ^= x >> 14;
  float u = (float)(x >> 8) * (1.f/16777216.f) + 1e-7f;
  return -__logf(-__logf(u));
}

// Scatter index into the packed A-fragment buffer for activation element (m, k).
// Layout: [mtile][kc][lane][e], lane = (k%32 >= 16 ? 16 : 0) + m%16,
// e = k%16   (contiguous k order, matching the packed B fragments).
__device__ __forceinline__ size_t afrag_idx(int m, int k){
  const int mtile = m >> 4, ml = m & 15;
  const int kc = k >> 5, kk = k & 31;
  const int lane = ((kk >> 4) << 4) + ml;
  const int e = kk & 15;    // contiguous: k = 16*(lane/16) + e
  return (((size_t)mtile * kKC + kc) * 32 + lane) * 16 + e;
}

// ---------------------------------------------------------------------------
// Pack the 4 LSTM weight matrices (4096x1024 f32) into bf16 WMMA B-fragments:
// B[k][n] = W[n][k]; layout [n_tile][k_chunk][lane][16] so each lane loads 32
// contiguous bytes per fragment.
// ---------------------------------------------------------------------------
__global__ __launch_bounds__(256) void pack_lstm_kernel(
    const float* __restrict__ m0, const float* __restrict__ m1,
    const float* __restrict__ m2, const float* __restrict__ m3,
    bf16* __restrict__ packed)
{
  const int blk = blockIdx.x;
  const int mat = blk >> 12;          // /4096
  const int row = blk & 4095;
  const float* src = (mat == 0) ? m0 : (mat == 1) ? m1 : (mat == 2) ? m2 : m3;
  bf16* dst = packed + (size_t)mat * kG * kH;
  const int ntile = row >> 4, nl = row & 15;
  for (int k = threadIdx.x; k < kH; k += 256){
    const int kc = k >> 5, kk = k & 31;
    const int lane = ((kk >= 16) ? 16 : 0) + nl;
    const int e = kk & 15;
    dst[(((size_t)ntile * kKC + kc) * 32 + lane) * 16 + e] =
        (bf16)src[(size_t)row * kH + k];
  }
}

// Weight-norm output matrix: Wn[n][k] = g[n] * v[n][k] / ||v[n]||, packed bf16.
__global__ __launch_bounds__(256) void pack_wn_kernel(
    const float* __restrict__ outv, const float* __restrict__ outg,
    bf16* __restrict__ wnp)
{
  __shared__ float red[256];
  const int row = blockIdx.x, tid = threadIdx.x;
  float s = 0.f;
  for (int k = tid; k < kH; k += 256){ float v = outv[(size_t)row*kH + k]; s += v*v; }
  red[tid] = s; __syncthreads();
  for (int st = 128; st > 0; st >>= 1){
    if (tid < st) red[tid] += red[tid + st];
    __syncthreads();
  }
  const float scale = outg[row] * rsqrtf(red[0] + 1e-30f);
  const int ntile = row >> 4, nl = row & 15;
  for (int k = tid; k < kH; k += 256){
    const int kc = k >> 5, kk = k & 31;
    const int lane = ((kk >= 16) ? 16 : 0) + nl;
    const int e = kk & 15;
    wnp[(((size_t)ntile * kKC + kc) * 32 + lane) * 16 + e] =
        (bf16)(outv[(size_t)row * kH + k] * scale);
  }
}

// hc0 = z @ z_proj_w^T + b ; scatter into h0/c0 (f32) and packed-A h0 (bf16).
__global__ __launch_bounds__(256) void init_hc_kernel(
    const float* __restrict__ z, const float* __restrict__ zw,
    const float* __restrict__ zb, float* __restrict__ h0,
    bf16* __restrict__ h0p, float* __restrict__ c)
{
  const int idx = blockIdx.x * 256 + threadIdx.x;   // 32*4096 total
  const int b = idx >> 12;
  const int n = idx & 4095;
  float s = zb[n];
  const float* zr = z + b * kZD;
  const float* wr = zw + (size_t)n * kZD;
  #pragma unroll 4
  for (int k = 0; k < kZD; ++k) s += zr[k] * wr[k];
  if (n < 2 * kH){
    const int l = n >> 10; const int col = n & 1023;
    h0[((size_t)l * kB + b) * kH + col] = s;
    h0p[(size_t)l * kAP + afrag_idx(b, col)] = (bf16)s;
  } else {
    const int n2 = n - 2 * kH; const int l = n2 >> 10;
    c[((size_t)l * kB + b) * kH + (n2 & 1023)] = s;
  }
}

// x0 = emb[BOS] broadcast (packed-A); bias[l] = b_ih[l] + b_hh[l].
__global__ __launch_bounds__(256) void init_misc_kernel(
    const float* __restrict__ emb,
    const float* __restrict__ bih0, const float* __restrict__ bhh0,
    const float* __restrict__ bih1, const float* __restrict__ bhh1,
    bf16* __restrict__ xap, float* __restrict__ bias)
{
  const int i = blockIdx.x * 256 + threadIdx.x;     // 40960 total
  if (i < kB * kH){
    const int b = i >> 10, h = i & 1023;
    xap[afrag_idx(b, h)] = (bf16)emb[kBOS * kH + h];
  } else {
    const int j = i - kB * kH;                      // < 8192
    const int l = j >> 12; const int n = j & 4095;
    bias[l * kG + n] = (l == 0) ? (bih0[n] + bhh0[n]) : (bih1[n] + bhh1[n]);
  }
}

// ---------------------------------------------------------------------------
// One LSTM cell layer. Block = 256 thr = 8 waves; wave = (gate, m-half).
// Block owns 16 hidden columns, computes i,f,g,o tiles and fuses nonlinearity.
// K-loop: pure vmem (A frag + B frag, both contiguous per lane) + WMMA.
// ---------------------------------------------------------------------------
__global__ __launch_bounds__(256) void lstm_cell_kernel(
    const bf16* __restrict__ xinp, const bf16* __restrict__ hprevp,
    float* __restrict__ hout, bf16* __restrict__ houtp,
    float* __restrict__ cst,
    const bf16* __restrict__ wih, const bf16* __restrict__ whh,
    const float* __restrict__ bias)
{
  __shared__ float gl[4][32][16];        // gate accumulators
  const int tid  = threadIdx.x;
  const int lane = tid & 31;
  const int wave = tid >> 5;
  const int gate = wave >> 1;
  const int m0   = (wave & 1) << 4;
  const int mtile = m0 >> 4;
  const int n0   = blockIdx.x << 4;
  const size_t ntile = (size_t)(((gate << 10) + n0) >> 4);

  v8f acc = {};
  const bf16* wmat[2] = { wih, whh };
  const bf16* amat[2] = { xinp, hprevp };

  for (int pass = 0; pass < 2; ++pass){
    const bf16* ap = amat[pass] + ((size_t)mtile * kKC * 32 + lane) * 16;
    const bf16* wp = wmat[pass] + ((ntile * kKC) * 32 + lane) * 16;
    #pragma unroll 4
    for (int kc = 0; kc < kKC; ++kc){
      if ((kc & 3) == 0)   // WGP-scope speculative prefetch 8 chunks ahead
        __builtin_prefetch((const void*)(wp + (size_t)(kc + 8) * 32 * 16), 0, 3);
      const v16bf a   = *(const v16bf*)(ap + (size_t)kc * 32 * 16);
      const v16bf bfr = *(const v16bf*)(wp + (size_t)kc * 32 * 16);
      acc = __builtin_amdgcn_wmma_f32_16x16x32_bf16(
                false, a, false, bfr, (short)0, acc, false, false);
    }
  }
  // spill accumulator (VGPR r: M = m0 + (lane<16 ? r : r+8), N = lane%16)
  #pragma unroll
  for (int r = 0; r < 8; ++r){
    const int m = m0 + ((lane < 16) ? r : r + 8);
    const int n = lane & 15;
    gl[gate][m][n] = acc[r] + bias[(gate << 10) + n0 + n];
  }
  __syncthreads();
  for (int i = tid; i < 32 * 16; i += 256){
    const int b = i >> 4, n = i & 15;
    const int col = n0 + n;
    const float ig = gl[0][b][n], fg = gl[1][b][n];
    const float gg = gl[2][b][n], og = gl[3][b][n];
    const float cprev = cst[(size_t)b * kH + col];
    const float c2 = sigf(fg) * cprev + sigf(ig) * tanhf(gg);
    const float hv = sigf(og) * tanhf(c2);
    cst[(size_t)b * kH + col]  = c2;
    hout[(size_t)b * kH + col] = hv;
    houtp[afrag_idx(b, col)]   = (bf16)hv;
  }
}

// LayerNorm -> packed-A bf16 fragments (sole consumer is the logits GEMM).
__global__ __launch_bounds__(256) void ln_kernel(
    const float* __restrict__ hin, const float* __restrict__ g,
    const float* __restrict__ bta, bf16* __restrict__ hidp)
{
  __shared__ float r1[256], r2[256];
  const int b = blockIdx.x, tid = threadIdx.x;
  float s = 0.f, s2 = 0.f;
  for (int i = tid; i < kH; i += 256){
    const float v = hin[(size_t)b * kH + i]; s += v; s2 += v * v;
  }
  r1[tid] = s; r2[tid] = s2; __syncthreads();
  for (int st = 128; st > 0; st >>= 1){
    if (tid < st){ r1[tid] += r1[tid + st]; r2[tid] += r2[tid + st]; }
    __syncthreads();
  }
  const float mean = r1[0] * (1.f / kH);
  const float var  = r2[0] * (1.f / kH) - mean * mean;
  const float rstd = rsqrtf(var + 1e-5f);
  for (int i = tid; i < kH; i += 256){
    const float v = (hin[(size_t)b * kH + i] - mean) * rstd * g[i] + bta[i];
    hidp[afrag_idx(b, i)] = (bf16)v;
  }
}

// ---------------------------------------------------------------------------
// Logits tile: block covers 64 vocab columns (8 waves = 2 m-halves x 4 n-tiles).
// ---------------------------------------------------------------------------
__global__ __launch_bounds__(256) void logits_kernel(
    const bf16* __restrict__ hidp, const bf16* __restrict__ wnp,
    const float* __restrict__ outb, float* __restrict__ dlog,
    float* __restrict__ pval, int* __restrict__ pidx, int t)
{
  __shared__ float nv[32][64];
  const int tid  = threadIdx.x;
  const int lane = tid & 31;
  const int wave = tid >> 5;
  const int m0   = (wave & 1) << 4;
  const int mtile = m0 >> 4;
  const int nt   = wave >> 1;                  // 0..3
  const int nbase = (blockIdx.x << 6) + (nt << 4);
  const size_t ntile = (size_t)(blockIdx.x * 4 + nt);

  v8f acc = {};
  const bf16* ap = hidp + ((size_t)mtile * kKC * 32 + lane) * 16;
  const bf16* wp = wnp + ((ntile * kKC) * 32 + lane) * 16;
  #pragma unroll 4
  for (int kc = 0; kc < kKC; ++kc){
    if ((kc & 3) == 0)   // WGP-scope speculative prefetch 8 chunks ahead
      __builtin_prefetch((const void*)(wp + (size_t)(kc + 8) * 32 * 16), 0, 3);
    const v16bf a   = *(const v16bf*)(ap + (size_t)kc * 32 * 16);
    const v16bf bfr = *(const v16bf*)(wp + (size_t)kc * 32 * 16);
    acc = __builtin_amdgcn_wmma_f32_16x16x32_bf16(
              false, a, false, bfr, (short)0, acc, false, false);
  }
  #pragma unroll
  for (int r = 0; r < 8; ++r){
    const int m = m0 + ((lane < 16) ? r : r + 8);
    const int n = lane & 15;
    const int col = nbase + n;
    const float logit = acc[r] + outb[col];
    dlog[((size_t)m * kL + t) * kV + col] = logit;
    const unsigned seed = ((unsigned)t * 0x9E3779B9u) ^ ((unsigned)m << 20) ^ (unsigned)col;
    nv[m][(nt << 4) + n] = logit + gumbelf(seed);
  }
  __syncthreads();
  if (tid < kB){
    float best = -3.4e38f; int bi = 0;
    for (int j = 0; j < 64; ++j){
      const float v = nv[tid][j];
      if (v > best){ best = v; bi = (blockIdx.x << 6) + j; }
    }
    pval[blockIdx.x * kB + tid] = best;
    pidx[blockIdx.x * kB + tid] = bi;
  }
}

// Reduce partial argmax -> token; gather embedding row (straight-through forward
// value is exactly one-hot) -> packed-A feedback buffer + embs output.
__global__ __launch_bounds__(256) void argmax_embed_kernel(
    const float* __restrict__ pval, const int* __restrict__ pidx,
    const float* __restrict__ emb, float* __restrict__ dtok,
    float* __restrict__ dembs, bf16* __restrict__ xap, int t)
{
  __shared__ int toks[kB];
  const int tid = threadIdx.x;
  if (tid < kB){
    float best = -3.4e38f; int bi = 0;
    for (int j = 0; j < kNB; ++j){
      const float v = pval[j * kB + tid];
      if (v > best){ best = v; bi = pidx[j * kB + tid]; }
    }
    toks[tid] = bi;
    dtok[tid * kL + t] = (float)bi;
  }
  __syncthreads();
  for (int i = tid; i < kB * kH; i += 256){
    const int b = i >> 10, h = i & 1023;
    const float e = emb[(size_t)toks[b] * kH + h];
    xap[afrag_idx(b, h)] = (bf16)e;
    dembs[((size_t)b * kL + t) * kH + h] = e;
  }
}

// ---------------------------------------------------------------------------
extern "C" void kernel_launch(void* const* d_in, const int* in_sizes, int n_in,
                              void* d_out, int out_size, void* d_ws, size_t ws_size,
                              hipStream_t stream)
{
  (void)in_sizes; (void)n_in; (void)out_size; (void)ws_size;
  const float* z    = (const float*)d_in[0];
  const float* zw   = (const float*)d_in[1];
  const float* zb   = (const float*)d_in[2];
  const float* emb  = (const float*)d_in[3];
  const float* wih0 = (const float*)d_in[4];
  const float* whh0 = (const float*)d_in[5];
  const float* bih0 = (const float*)d_in[6];
  const float* bhh0 = (const float*)d_in[7];
  const float* wih1 = (const float*)d_in[8];
  const float* whh1 = (const float*)d_in[9];
  const float* bih1 = (const float*)d_in[10];
  const float* bhh1 = (const float*)d_in[11];
  const float* lng  = (const float*)d_in[12];
  const float* lnb  = (const float*)d_in[13];
  const float* outv = (const float*)d_in[14];
  const float* outg = (const float*)d_in[15];
  const float* outb = (const float*)d_in[16];

  // workspace carve-out (~100.4 MB): all weights bf16 -> L2-resident
  char* w = (char*)d_ws;
  size_t off = 0;
  auto take = [&](size_t bytes) -> char* {
    char* p = w + off; off += (bytes + 255) & ~(size_t)255; return p;
  };
  bf16*  wnp  = (bf16*) take((size_t)kV * kH * 2);        // packed weight-norm matrix
  bf16*  lp   = (bf16*) take((size_t)4 * kG * kH * 2);    // packed wih0,whh0,wih1,whh1
  float* bias = (float*)take((size_t)2 * kG * 4);         // fused b_ih + b_hh
  float* hpp  = (float*)take((size_t)2 * 2 * kB * kH * 4);// f32 ping-pong [pp][layer][B][H]
  bf16*  hpk  = (bf16*) take((size_t)2 * 2 * kAP * 2);    // packed-A ping-pong h
  float* cst  = (float*)take((size_t)2 * kB * kH * 4);    // cell state [layer][B][H]
  bf16*  xap  = (bf16*) take((size_t)kAP * 2);            // packed-A token embedding
  bf16*  hidp = (bf16*) take((size_t)kAP * 2);            // packed-A post-LN hidden
  float* pval = (float*)take((size_t)kNB * kB * 4);       // partial argmax values
  int*   pidx = (int*)  take((size_t)kNB * kB * 4);       // partial argmax indices

  float* dtok  = (float*)d_out;                           // (B, L)
  float* dlog  = dtok + (size_t)kB * kL;                  // (B, L, V)
  float* dembs = dlog + (size_t)kB * kL * kV;             // (B, L, H)

  // one-time prep: bf16 packing + initial state
  pack_lstm_kernel<<<dim3(4 * 4096), dim3(256), 0, stream>>>(wih0, whh0, wih1, whh1, lp);
  pack_wn_kernel  <<<dim3(kV),       dim3(256), 0, stream>>>(outv, outg, wnp);
  init_hc_kernel  <<<dim3(512),      dim3(256), 0, stream>>>(z, zw, zb, hpp, hpk, cst);
  init_misc_kernel<<<dim3(160),      dim3(256), 0, stream>>>(emb, bih0, bhh0, bih1, bhh1, xap, bias);

  const size_t MSZ = (size_t)kG * kH;   // one packed LSTM matrix, in elements
  for (int t = 0; t < kL; ++t){
    const int p = t & 1;
    float* h0n = hpp + (size_t)(1 - p) * 2 * kB * kH;
    float* h1n = h0n + (size_t)kB * kH;
    bf16* h0pc = hpk + (size_t)p       * 2 * kAP;
    bf16* h1pc = h0pc + (size_t)kAP;
    bf16* h0pn = hpk + (size_t)(1 - p) * 2 * kAP;
    bf16* h1pn = h0pn + (size_t)kAP;
    lstm_cell_kernel<<<dim3(kH / 16), dim3(256), 0, stream>>>(
        xap,  h0pc, h0n, h0pn, cst,                  lp,           lp + MSZ,     bias);
    lstm_cell_kernel<<<dim3(kH / 16), dim3(256), 0, stream>>>(
        h0pn, h1pc, h1n, h1pn, cst + (size_t)kB*kH,  lp + 2 * MSZ, lp + 3 * MSZ, bias + kG);
    ln_kernel<<<dim3(kB), dim3(256), 0, stream>>>(h1n, lng, lnb, hidp);
    logits_kernel<<<dim3(kNB), dim3(256), 0, stream>>>(hidp, wnp, outb, dlog, pval, pidx, t);
    argmax_embed_kernel<<<dim3(1), dim3(256), 0, stream>>>(pval, pidx, emb, dtok, dembs, xap, t);
  }
}
